// MsaHmmCell_88244398063824
// MI455X (gfx1250) — matrix operand order repaired and checked by reading, not compiled, not run on hardware
//
#include <hip/hip_runtime.h>

// ---------------------------------------------------------------------------
// HMM forward (MsaHmmCell) for MI455X / gfx1250.
// Dominant cost: 511 x (256x1027)@(1027x1027) GEMM -> bf16 WMMA, f32 accum.
// A matrix softmaxed + transposed to bf16 once (L2-resident, 2.2 MB).
// Output forward probs written once to HBM (538 MB), normalized in place.
// ---------------------------------------------------------------------------

#define QN     1027     // number of states
#define QP     1056     // padded to multiple of 32 (K) and 16 (N)
#define ALPHAN 26
#define BATCH  256
#define TLEN   512
#define NTILES 65       // ceil(1027/16) -> covers n in [0,1040)
#define MTILES 16       // 256/16
#define WPB    8        // waves per block
#define GEMM_BLOCKS ((MTILES * NTILES) / WPB)   // 1040/8 = 130 exactly

typedef __attribute__((ext_vector_type(16))) __bf16 v16bf;
typedef __attribute__((ext_vector_type(8)))  __bf16 v8bf;
typedef __attribute__((ext_vector_type(8)))  float  v8f;

// ---------------- small helpers ----------------

__device__ __forceinline__ float blockReduceSum(float v, float* red) {
    int tid = threadIdx.x;
    red[tid] = v;
    __syncthreads();
    for (int s = 128; s > 0; s >>= 1) {
        if (tid < s) red[tid] += red[tid + s];
        __syncthreads();
    }
    float r = red[0];
    __syncthreads();
    return r;
}

__device__ __forceinline__ float blockReduceMax(float v, float* red) {
    int tid = threadIdx.x;
    red[tid] = v;
    __syncthreads();
    for (int s = 128; s > 0; s >>= 1) {
        if (tid < s) red[tid] = fmaxf(red[tid], red[tid + s]);
        __syncthreads();
    }
    float r = red[0];
    __syncthreads();
    return r;
}

// ---------------- setup kernels ----------------

__global__ void fill0_kernel(unsigned int* p, int n) {
    int i = blockIdx.x * 256 + threadIdx.x;
    if (i < n) p[i] = 0u;
}

// Row softmax of A_logits (QN x QN), written TRANSPOSED as bf16 into AT
// (QP x QP, zero padded): AT[n*QP + k] = softmax(A_logits[k,:])[n].
__global__ void softmaxA_kernel(const float* __restrict__ Alog,
                                __bf16* __restrict__ AT) {
    __shared__ float red[256];
    int k = blockIdx.x;                      // row of A
    const float* row = Alog + (size_t)k * QN;
    float mx = -__builtin_inff();
    for (int n = threadIdx.x; n < QN; n += 256) mx = fmaxf(mx, row[n]);
    mx = blockReduceMax(mx, red);
    float s = 0.f;
    for (int n = threadIdx.x; n < QN; n += 256) s += __expf(row[n] - mx);
    s = blockReduceSum(s, red);
    float inv = 1.f / s;
    for (int n = threadIdx.x; n < QN; n += 256)
        AT[(size_t)n * QP + k] = (__bf16)(__expf(row[n] - mx) * inv);
}

// Row softmax of B_logits (QN x 26), written transposed: BT[a*QP + q].
__global__ void softmaxB_kernel(const float* __restrict__ Blog,
                                float* __restrict__ BT) {
    int q = blockIdx.x;
    int l = threadIdx.x;                     // 32 lanes, 26 active
    float v = (l < ALPHAN) ? Blog[(size_t)q * ALPHAN + l] : -__builtin_inff();
    float mx = v;
    for (int off = 16; off > 0; off >>= 1) mx = fmaxf(mx, __shfl_xor(mx, off, 32));
    float e = (l < ALPHAN) ? __expf(v - mx) : 0.f;
    float s = e;
    for (int off = 16; off > 0; off >>= 1) s += __shfl_xor(s, off, 32);
    if (l < ALPHAN) BT[(size_t)l * QP + q] = e / s;
}

__global__ void softmaxInit_kernel(const float* __restrict__ ilog,
                                   float* __restrict__ ism) {
    __shared__ float red[256];
    float mx = -__builtin_inff();
    for (int q = threadIdx.x; q < QN; q += 256) mx = fmaxf(mx, ilog[q]);
    mx = blockReduceMax(mx, red);
    float s = 0.f;
    for (int q = threadIdx.x; q < QN; q += 256) s += __expf(ilog[q] - mx);
    s = blockReduceSum(s, red);
    float inv = 1.f / s;
    for (int q = threadIdx.x; q < QN; q += 256)
        ism[q] = __expf(ilog[q] - mx) * inv;
}

// ---------------- t = 0 (init branch) ----------------

__global__ void step0_kernel(const int* __restrict__ x,
                             const float* __restrict__ BT,
                             const float* __restrict__ ism,
                             float* __restrict__ out,
                             __bf16* __restrict__ fbf,
                             float* __restrict__ ll) {
    __shared__ float red[256];
    int b = blockIdx.x;
    int a = x[(size_t)b * TLEN];             // x[b, 0]
    float* base = out + (size_t)b * TLEN * QN;   // forward[b, 0, :]
    const float* brow = BT + (size_t)a * QP;
    float vloc[5];
    float s = 0.f;
    int i = 0;
    for (int q = threadIdx.x; q < QN; q += 256, ++i) {
        float v = brow[q] * ism[q];
        vloc[i] = v;
        s += v;
    }
    float S = blockReduceSum(s, red);
    if (threadIdx.x == 0) ll[b] = logf(S);
    float inv = 1.f / S;
    i = 0;
    for (int q = threadIdx.x; q < QN; q += 256, ++i) {
        float v = vloc[i] * inv;
        base[q] = v;
        fbf[(size_t)b * QP + q] = (__bf16)v;
    }
    // zero K padding of the state once; never touched again
    for (int q = QN + threadIdx.x; q < QP; q += 256)
        fbf[(size_t)b * QP + q] = (__bf16)0.f;
}

// ---------------- per-step WMMA GEMM: raw = f @ A ----------------
// 1040 waves, one 16x16 output tile each, K = 1056 in 33 chunks of 32.
// A-operand layout (16-bit A 16x32): lane<16 -> M=lane, K {0..7,16..23};
//                                    lane>=16 -> M=lane-16, K {8..15,24..31}.
// B-operand layout (16-bit B 32x16): lane<16 -> N=lane, K {0..7,16..23};
//                                    lane>=16 -> N=lane-16, K {8..15,24..31}.
// AT is pre-transposed, so both operands load as contiguous 16B chunks.
__global__ void gemm_step_kernel(const __bf16* __restrict__ f,
                                 const __bf16* __restrict__ AT,
                                 float* __restrict__ out,
                                 int t) {
    int wave = (blockIdx.x * blockDim.x + threadIdx.x) >> 5;
    int lane = threadIdx.x & 31;
    int mTile = wave / NTILES;
    int nTile = wave % NTILES;
    int lid   = lane & 15;
    int m     = mTile * 16 + lid;
    int n     = nTile * 16 + lid;
    int aoff  = (lane < 16) ? 0 : 8;
    int boff  = (lane < 16) ? 0 : 8;
    const __bf16* frow = f  + (size_t)m * QP;
    const __bf16* brow = AT + (size_t)n * QP;

    v8f acc = {};
    for (int k0 = 0; k0 < QP; k0 += 32) {
        v8bf al = *(const v8bf*)(frow + k0 + aoff);
        v8bf ah = *(const v8bf*)(frow + k0 + aoff + 16);
        v8bf bl = *(const v8bf*)(brow + k0 + boff);
        v8bf bh = *(const v8bf*)(brow + k0 + boff + 16);
        v16bf av, bv;
#pragma unroll
        for (int j = 0; j < 8; ++j) {
            av[j] = al[j]; av[j + 8] = ah[j];
            bv[j] = bl[j]; bv[j + 8] = bh[j];
        }
        acc = __builtin_amdgcn_wmma_f32_16x16x32_bf16(
            /*neg_a=*/false, av, /*neg_b=*/false, bv,
            /*c_mod=*/(short)0, acc, /*reuse_a=*/false, /*reuse_b=*/false);
    }

    if (n < QN) {
        int mBase = mTile * 16 + ((lane < 16) ? 0 : 8);
#pragma unroll
        for (int r = 0; r < 8; ++r) {
            int b = mBase + r;   // C/D layout: VGPR r -> M = r (+8 for hi lanes)
            out[(size_t)b * TLEN * QN + (size_t)t * QN + n] = acc[r];
        }
    }
}

// ---------------- per-step normalize: *= E, scale, log-lik ----------------

__global__ void norm_step_kernel(const int* __restrict__ x,
                                 const float* __restrict__ BT,
                                 float* __restrict__ out,
                                 __bf16* __restrict__ fbf,
                                 float* __restrict__ ll,
                                 int t) {
    __shared__ float red[256];
    int b = blockIdx.x;
    int a = x[(size_t)b * TLEN + t];
    float* base = out + (size_t)b * TLEN * QN + (size_t)t * QN;
    const float* brow = BT + (size_t)a * QP;
    float vloc[5];
    float s = 0.f;
    int i = 0;
    for (int q = threadIdx.x; q < QN; q += 256, ++i) {
        float v = base[q] * brow[q];         // emit * (f @ A)
        vloc[i] = v;
        s += v;
    }
    float S = blockReduceSum(s, red);
    if (threadIdx.x == 0) ll[b] += logf(S);
    float inv = 1.f / S;
    i = 0;
    for (int q = threadIdx.x; q < QN; q += 256, ++i) {
        float v = vloc[i] * inv;
        base[q] = v;
        fbf[(size_t)b * QP + q] = (__bf16)v;
    }
}

__global__ void copy_ll_kernel(const float* __restrict__ ll,
                               float* __restrict__ out) {
    out[threadIdx.x] = ll[threadIdx.x];
}

// ---------------- host launcher ----------------

extern "C" void kernel_launch(void* const* d_in, const int* in_sizes, int n_in,
                              void* d_out, int out_size, void* d_ws, size_t ws_size,
                              hipStream_t stream) {
    const int*   x      = (const int*)  d_in[0];
    const float* Alog   = (const float*)d_in[1];
    const float* Blog   = (const float*)d_in[2];
    const float* initlg = (const float*)d_in[3];
    float*       out    = (float*)d_out;
    char*        ws     = (char*)d_ws;

    // workspace layout (all 16B-aligned)
    const size_t szAT   = (size_t)QP * QP * sizeof(__bf16);   // 2,230,272
    const size_t szBT   = (size_t)ALPHAN * QP * sizeof(float);//   109,824
    const size_t szInit = (size_t)QP * sizeof(float);         //     4,224
    const size_t offAT   = 0;
    const size_t offBT   = offAT + szAT;
    const size_t offInit = offBT + szBT;
    const size_t offF    = offInit + szInit;                  // bf16 state
    const size_t szF     = (size_t)BATCH * QP * sizeof(__bf16);
    const size_t offLL   = offF + szF;

    __bf16* AT  = (__bf16*)(ws + offAT);
    float*  BT  = (float*) (ws + offBT);
    float*  ism = (float*) (ws + offInit);
    __bf16* fbf = (__bf16*)(ws + offF);
    float*  ll  = (float*) (ws + offLL);

    // zero AT/BT/init padding region (contiguous block)
    int zeroWords = (int)((szAT + szBT + szInit) / 4);
    fill0_kernel<<<(zeroWords + 255) / 256, 256, 0, stream>>>(
        (unsigned int*)ws, zeroWords);

    // softmaxes
    softmaxA_kernel<<<QN, 256, 0, stream>>>(Alog, AT);
    softmaxB_kernel<<<QN, 32, 0, stream>>>(Blog, BT);
    softmaxInit_kernel<<<1, 256, 0, stream>>>(initlg, ism);

    // t = 0: forward = E0 * init, normalize
    step0_kernel<<<BATCH, 256, 0, stream>>>(x, BT, ism, out, fbf, ll);

    // t = 1..T-1: WMMA GEMM then emission/normalize
    for (int t = 1; t < TLEN; ++t) {
        gemm_step_kernel<<<GEMM_BLOCKS, WPB * 32, 0, stream>>>(fbf, AT, out, t);
        norm_step_kernel<<<BATCH, 256, 0, stream>>>(x, BT, out, fbf, ll, t);
    }

    // append loglik after forward in d_out
    copy_ll_kernel<<<1, BATCH, 0, stream>>>(
        ll, out + (size_t)BATCH * TLEN * QN);
}
